// DRPO_16604343566649
// MI455X (gfx1250) — matrix operand order repaired and checked by reading, not compiled, not run on hardware
//
#include <hip/hip_runtime.h>
#include <stdint.h>

// ---------------- problem constants ----------------
#define HSZ   128
#define SNS   32
#define FNS   16
#define BSZ   128
#define TSS   512
#define MROWS (BSZ*SNS)     // 4096  (rows of LSTM GEMM)
#define NGATE (4*HSZ)       // 512
#define NL1   (HSZ*7)       // 896
#define KMLP  (SNS*HSZ)     // 4096

// ---------------- WMMA types ----------------
typedef __attribute__((ext_vector_type(16))) __bf16        v16bf;
typedef __attribute__((ext_vector_type(16))) unsigned short v16us;
typedef __attribute__((ext_vector_type(8)))  float          v8f;
typedef __attribute__((ext_vector_type(4)))  unsigned int   v4u;

union Frag { v16bf b; v16us us; v4u q[2]; };

__device__ __forceinline__ unsigned short f2bf(float f) {
  unsigned int u = __float_as_uint(f);
  return (unsigned short)((u + 0x7FFFu + ((u >> 16) & 1u)) >> 16);
}
__device__ __forceinline__ float sigm(float x) { return 1.f / (1.f + __expf(-x)); }
__device__ __forceinline__ float wsum(float v) {
  #pragma unroll
  for (int o = 16; o > 0; o >>= 1) v += __shfl_xor(v, o, 32);
  return v;
}
__device__ __forceinline__ float wmaxf(float v) {
  #pragma unroll
  for (int o = 16; o > 0; o >>= 1) v = fmaxf(v, __shfl_xor(v, o, 32));
  return v;
}

// A-matrix 16x32 bf16 frag: lanes 0-15 hold K={c..c+7, c+16..c+23}, c = (lane&16)?8:0 (+k0)
__device__ __forceinline__ void load_fragA(const unsigned short* row, int k0, int lane, Frag& f) {
  int c = k0 + ((lane & 16) ? 8 : 0);
  f.q[0] = *(const v4u*)(row + c);
  f.q[1] = *(const v4u*)(row + c + 16);
}
// B-matrix 32x16 bf16 frag (Wt stored [N][K] row-major): lane holds column n=lane&15,
// K={c..c+7, c+16..c+23}: c = (lane&16)?8:0 (+k0)
__device__ __forceinline__ void load_fragB(const unsigned short* row, int k0, int lane, Frag& f) {
  int c = k0 + ((lane & 16) ? 8 : 0);
  f.q[0] = *(const v4u*)(row + c);
  f.q[1] = *(const v4u*)(row + c + 16);
}
#define WMMA_BF16(A, B, C) \
  __builtin_amdgcn_wmma_f32_16x16x32_bf16(false, (A), false, (B), (short)0, (C), false, false)

// ---------------- prep kernels ----------------
__global__ void k_prep_wih(const float* __restrict__ W, unsigned short* __restrict__ o) {
  int i = blockIdx.x * blockDim.x + threadIdx.x;            // 512*32
  if (i >= NGATE * 32) return;
  int n = i >> 5, k = i & 31;
  o[i] = (k < FNS + 1) ? f2bf(W[n * (FNS + 1) + k]) : (unsigned short)0;
}
__global__ void k_prep_whh(const float* __restrict__ W, unsigned short* __restrict__ o) {
  int i = blockIdx.x * blockDim.x + threadIdx.x;            // 512*128
  if (i >= NGATE * HSZ) return;
  o[i] = f2bf(W[i]);
}
__global__ void k_prep_w1t(const float* __restrict__ W, unsigned short* __restrict__ o) {
  int i = blockIdx.x * blockDim.x + threadIdx.x;            // 896*4096
  if (i >= NL1 * KMLP) return;
  int n = i / KMLP, k = i % KMLP;
  o[i] = f2bf(W[(size_t)k * NL1 + n]);                      // W1 is [K][N]
}
__global__ void k_prep_w2t(const float* __restrict__ W, unsigned short* __restrict__ o) {
  int i = blockIdx.x * blockDim.x + threadIdx.x;            // 128*896
  if (i >= HSZ * NL1) return;
  int n = i / NL1, k = i % NL1;
  o[i] = f2bf(W[(size_t)k * HSZ + n]);                      // W2 is [K][N]
}
__global__ void k_init_state(unsigned short* __restrict__ hbf, float* __restrict__ cbuf,
                             float* __restrict__ st, float* __restrict__ om,
                             float* __restrict__ cash) {
  int i = blockIdx.x * blockDim.x + threadIdx.x;            // 4096*128
  if (i >= MROWS * HSZ) return;
  hbf[i] = 0x3F80;                                          // bf16(1.0)
  cbuf[i] = 1.f;
  if (i < BSZ * SNS) { st[i] = 100.f; om[i] = 1.f / 32.f; } // softmax(const) = 1/32
  if (i < BSZ) { cash[i] = 0.f; cash[BSZ + i] = 1000000.f; }
}

// ---------------- per-step kernel A: fused LSTM cell ----------------
// g = [x_t | omega] @ W_ih^T + h @ W_hh^T + b ; gates -> (h2, c2)
// wave = one 16x16 (m, hs) tile computing all 4 gate tiles.
// All 4 gate B-frags live simultaneously so waits stagger (0x6/0x4/0x2/0x0);
// next-K A-frag preloaded during the current 4 WMMAs.
__global__ __launch_bounds__(128) void k_lstm_step(
    const float* __restrict__ inputs, const unsigned short* __restrict__ Wih,
    const unsigned short* __restrict__ Whh, const float* __restrict__ b_ih,
    const float* __restrict__ b_hh, const unsigned short* __restrict__ h_in,
    unsigned short* __restrict__ h_out, float* __restrict__ c_buf,
    const float* __restrict__ omega, int t) {
  int lane = threadIdx.x & 31;
  int wave = blockIdx.x * (blockDim.x >> 5) + (threadIdx.x >> 5);   // 0..2047
  int mTile = wave >> 3;                                            // 0..255
  int hTile = wave & 7;                                             // 0..7
  int m0 = mTile * 16;
  int hcol = hTile * 16 + (lane & 15);                              // hs column 0..127

  v8f acc[4];
  #pragma unroll
  for (int g = 0; g < 4; ++g)
    #pragma unroll
    for (int r = 0; r < 8; ++r) acc[g][r] = 0.f;

  const unsigned short* arow = h_in + (size_t)(m0 + (lane & 15)) * HSZ;
  const unsigned short* w0 = Whh + (size_t)(0 * HSZ + hcol) * HSZ;
  const unsigned short* w1 = Whh + (size_t)(1 * HSZ + hcol) * HSZ;
  const unsigned short* w2 = Whh + (size_t)(2 * HSZ + hcol) * HSZ;
  const unsigned short* w3 = Whh + (size_t)(3 * HSZ + hcol) * HSZ;

  // K-block 0: xin (17-wide, zero-padded to 32) x W_ih^T
  {
    Frag a;
    int m = m0 + (lane & 15);
    int bb = m >> 5, ss = m & 31;
    int c = (lane & 16) ? 8 : 0;
    const float* xp = inputs + (((size_t)bb * TSS + t) * SNS + ss) * FNS;
    #pragma unroll
    for (int j = 0; j < 8; ++j) a.us[j] = f2bf(xp[c + j]);          // k = c+j < 16
    #pragma unroll
    for (int j = 0; j < 8; ++j) {
      int k = c + 16 + j;
      a.us[8 + j] = (k == 16) ? f2bf(omega[m]) : (unsigned short)0;
    }
    Frag g0, g1, g2, g3;
    load_fragB(Wih + (size_t)(0 * HSZ + hcol) * 32, 0, lane, g0);
    load_fragB(Wih + (size_t)(1 * HSZ + hcol) * 32, 0, lane, g1);
    load_fragB(Wih + (size_t)(2 * HSZ + hcol) * 32, 0, lane, g2);
    load_fragB(Wih + (size_t)(3 * HSZ + hcol) * 32, 0, lane, g3);
    acc[0] = WMMA_BF16(a.b, g0.b, acc[0]);
    acc[1] = WMMA_BF16(a.b, g1.b, acc[1]);
    acc[2] = WMMA_BF16(a.b, g2.b, acc[2]);
    acc[3] = WMMA_BF16(a.b, g3.b, acc[3]);
  }
  // K-blocks over h_in x W_hh^T (K = 128)
  Frag ah; load_fragA(arow, 0, lane, ah);
  #pragma unroll
  for (int kb = 0; kb < 4; ++kb) {
    int k0 = kb * 32;
    Frag g0, g1, g2, g3;                       // all 4 loads issued before any WMMA
    load_fragB(w0, k0, lane, g0);
    load_fragB(w1, k0, lane, g1);
    load_fragB(w2, k0, lane, g2);
    load_fragB(w3, k0, lane, g3);
    Frag an = ah;
    if (kb < 3) load_fragA(arow, k0 + 32, lane, an);
    acc[0] = WMMA_BF16(ah.b, g0.b, acc[0]);
    acc[1] = WMMA_BF16(ah.b, g1.b, acc[1]);
    acc[2] = WMMA_BF16(ah.b, g2.b, acc[2]);
    acc[3] = WMMA_BF16(ah.b, g3.b, acc[3]);
    ah = an;
  }
  // epilogue: gates -> c2, h2
  float bi = b_ih[0 * HSZ + hcol] + b_hh[0 * HSZ + hcol];
  float bff = b_ih[1 * HSZ + hcol] + b_hh[1 * HSZ + hcol];
  float bg = b_ih[2 * HSZ + hcol] + b_hh[2 * HSZ + hcol];
  float bo = b_ih[3 * HSZ + hcol] + b_hh[3 * HSZ + hcol];
  int rbase = m0 + ((lane & 16) ? 8 : 0);
  #pragma unroll
  for (int r = 0; r < 8; ++r) {
    int m = rbase + r;
    float iv = acc[0][r] + bi;
    float fv = acc[1][r] + bff;
    float gv = acc[2][r] + bg;
    float ov = acc[3][r] + bo;
    float co = c_buf[(size_t)m * HSZ + hcol];
    float cn = sigm(fv) * co + sigm(iv) * tanhf(gv);
    float hn = sigm(ov) * tanhf(cn);
    c_buf[(size_t)m * HSZ + hcol] = cn;
    h_out[(size_t)m * HSZ + hcol] = f2bf(hn);
  }
}

// ---- K-split, software-pipelined GEMM body shared by MLP L1/L2 ----
// Block = 1 output tile (16x16); 4 waves split K; LDS reduce; wave 0 does epilogue.
template <int KTOT, int LDA_N, bool OUT_BF16>
__device__ __forceinline__ void gemm_tile_ksplit(
    const unsigned short* __restrict__ A, const unsigned short* __restrict__ Bt,
    const float* __restrict__ bias, void* __restrict__ out,
    int m0, int n0) {
  int lane = threadIdx.x & 31;
  int wv = threadIdx.x >> 5;                 // 0..3 K-split segment
  constexpr int KSEG = KTOT / 4;
  int kbase = wv * KSEG;
  int ncol = n0 + (lane & 15);

  const unsigned short* arow = A + (size_t)(m0 + (lane & 15)) * KTOT;
  const unsigned short* brow = Bt + (size_t)ncol * KTOT;

  v8f acc;
  #pragma unroll
  for (int r = 0; r < 8; ++r) acc[r] = 0.f;

  // software pipeline: frags for k+32 in flight while WMMA consumes k
  Frag a_cur, b_cur;
  load_fragA(arow, kbase, lane, a_cur);
  load_fragB(brow, kbase, lane, b_cur);
  for (int k0 = 0; k0 < KSEG - 32; k0 += 32) {
    Frag a_nxt, b_nxt;
    load_fragA(arow, kbase + k0 + 32, lane, a_nxt);
    load_fragB(brow, kbase + k0 + 32, lane, b_nxt);
    acc = WMMA_BF16(a_cur.b, b_cur.b, acc);
    a_cur = a_nxt; b_cur = b_nxt;
  }
  acc = WMMA_BF16(a_cur.b, b_cur.b, acc);

  // LDS reduction of the 4 K-partials
  __shared__ float red[3][32][8];
  if (wv != 0) {
    #pragma unroll
    for (int r = 0; r < 8; ++r) red[wv - 1][lane][r] = acc[r];
  }
  __syncthreads();
  if (wv == 0) {
    float bs = bias[ncol];
    int rbase = m0 + ((lane & 16) ? 8 : 0);
    #pragma unroll
    for (int r = 0; r < 8; ++r) {
      float v = acc[r] + red[0][lane][r] + red[1][lane][r] + red[2][lane][r] + bs;
      v = v * sigm(v);                       // silu
      if (OUT_BF16)
        ((unsigned short*)out)[(size_t)(rbase + r) * LDA_N + ncol] = f2bf(v);
      else
        ((float*)out)[(size_t)(rbase + r) * LDA_N + ncol] = v;
    }
  }
}

// MLP layer 1: (128 x 4096) @ (4096 x 896) + b1, silu -> bf16. 448 blocks.
__global__ __launch_bounds__(128) void k_mlp1(
    const unsigned short* __restrict__ h2, const unsigned short* __restrict__ W1t,
    const float* __restrict__ b1, unsigned short* __restrict__ x1) {
  int tile = blockIdx.x;                     // 8 x 56 tiles
  gemm_tile_ksplit<KMLP, NL1, true>(h2, W1t, b1, x1, (tile / 56) * 16, (tile % 56) * 16);
}
// MLP layer 2: (128 x 896) @ (896 x 128) + b2, silu -> f32. 64 blocks.
__global__ __launch_bounds__(128) void k_mlp2(
    const unsigned short* __restrict__ x1, const unsigned short* __restrict__ W2t,
    const float* __restrict__ b2, float* __restrict__ x2) {
  int tile = blockIdx.x;                     // 8 x 8 tiles
  gemm_tile_ksplit<NL1, HSZ, false>(x1, W2t, b2, x2, (tile >> 3) * 16, (tile & 7) * 16);
}

// ---------------- per-step kernel D: MLP L3/L4 + trading (1 wave per batch) -----------
__global__ __launch_bounds__(32) void k_head_trade(
    const float* __restrict__ x2, const float* __restrict__ W3, const float* __restrict__ b3,
    const float* __restrict__ W4, const float* __restrict__ b4,
    const float* __restrict__ bp_all, const float* __restrict__ sp_all,
    float* __restrict__ st, float* __restrict__ omega, float* __restrict__ cash,
    float* __restrict__ asset_raw, float* __restrict__ out_ts, int t) {
  int b = blockIdx.x, lane = threadIdx.x;
  __shared__ float xs[HSZ];
  xs[lane]      = x2[(size_t)b * HSZ + lane];
  xs[lane + 32] = x2[(size_t)b * HSZ + lane + 32];
  xs[lane + 64] = x2[(size_t)b * HSZ + lane + 64];
  xs[lane + 96] = x2[(size_t)b * HSZ + lane + 96];
  __syncthreads();
  // layer3: 64 outputs, lane owns j=lane and j=lane+32
  float o3a = b3[lane], o3b = b3[lane + 32];
  for (int k = 0; k < HSZ; ++k) {
    float xv = xs[k];
    o3a += xv * W3[k * 64 + lane];
    o3b += xv * W3[k * 64 + lane + 32];
  }
  o3a = o3a * sigm(o3a);
  o3b = o3b * sigm(o3b);
  // layer4 -> scalar out (all lanes)
  float outv = wsum(o3a * W4[lane] + o3b * W4[lane + 32]) + b4[0];
  if (lane == 0) out_ts[(size_t)b * TSS + t] = outv;

  // trading step (out has last-dim 1 -> oc = out - mean(out) = 0, computed faithfully)
  float oc = outv - outv;
  float sts = st[b * SNS + lane];
  float ns = sts + oc; if (ns <= 0.f) ns = 0.f;
  float wc = ns - sts;
  float sell_a = (wc >= 0.f) ? 0.f : wc;
  float buy_a  = (wc <= 0.f) ? 0.f : wc;
  int tp = (t + TSS - 1) % TSS;                       // roll(.,1) -> index t-1 (wrap)
  float bpv = bp_all[((size_t)b * TSS + tp) * SNS + lane];
  float spv = sp_all[((size_t)b * TSS + tp) * SNS + lane];
  float mid = (bpv + spv) * 0.5f;
  float bpt = mid + (bpv - mid);
  float spt = mid + (spv - mid);
  float buy_costs  = -buy_a * bpt * 0.001f;
  float sell_costs =  sell_a * spt * 0.001f;
  float sell_amt = wsum(-sell_a * spt + sell_costs);
  float this_cash = cash[b] + sell_amt;
  float buy_amt = wsum(buy_a * bpt - buy_costs);
  float factor = this_cash / buy_amt;
  if (!(fabsf(factor) < 3.4e38f)) factor = 0.f;       // !isfinite -> 0
  buy_a *= factor; buy_amt *= factor;
  float next_cash = this_cash - buy_amt;
  float nst = sts + sell_a + buy_a;
  float hold = wsum(nst * mid);
  float asset_t = next_cash + hold;
  float mx = wmaxf(nst);
  float e = __expf(nst - mx);
  float se = wsum(e);
  st[b * SNS + lane] = nst;
  omega[b * SNS + lane] = e / se;
  if (lane == 0) {
    float ci = cash[BSZ + b];
    cash[b] = ci;                                     // c_im1' = c_i
    cash[BSZ + b] = next_cash;                        // c_i'   = next_cash
    asset_raw[(size_t)b * (TSS + 1) + t + 1] = asset_t;
  }
}

// ---------------- post kernel: returns, discounted E, loss/profit/reward --------------
#define LOSS_OFF   (BSZ*TSS)
#define PROFIT_OFF (LOSS_OFF+1)
#define AR_OFF     (LOSS_OFF+2)
#define ASSET_OFF  (LOSS_OFF+3)

__global__ __launch_bounds__(128) void k_post(
    const float* __restrict__ bp_all, const float* __restrict__ sp_all,
    float* __restrict__ asset_raw, float* __restrict__ r_buf, float* __restrict__ E_buf,
    float* __restrict__ dout) {
  int b = threadIdx.x;
  // asset0 = TOTAL_MONEY + sum_s(100 * mid[b,0,s])
  float h0 = 0.f;
  for (int s = 0; s < SNS; ++s) {
    float bpv = bp_all[(size_t)b * TSS * SNS + s];
    float spv = sp_all[(size_t)b * TSS * SNS + s];
    h0 += 100.f * ((bpv + spv) * 0.5f);
  }
  float a0 = 1000000.f + h0;
  asset_raw[(size_t)b * (TSS + 1)] = a0;
  float* aout = dout + ASSET_OFF + (size_t)b * (TSS + 1);
  aout[0] = 1.f;                                       // a0/a0
  float prev = 1.f, sum_r = 0.f;
  for (int tt = 1; tt <= TSS; ++tt) {
    float an = asset_raw[(size_t)b * (TSS + 1) + tt] / a0;
    aout[tt] = an;
    float r = an - prev;                               // (a[t]-a[t-1]) / a[0], a[0]==1
    r_buf[(size_t)b * TSS + tt - 1] = r;
    sum_r += r;
    prev = an;
  }
  float El = aout[TSS];                                // E_last = a[-1]/a[0]
  E_buf[(size_t)b * TSS + TSS - 1] = El;
  float e = El;
  for (int tt = TSS - 2; tt >= 0; --tt) {
    e = r_buf[(size_t)b * TSS + tt] + 0.99f * e;
    E_buf[(size_t)b * TSS + tt] = e;
  }
  float E0 = e;                                        // E_pi[b,0]
  // V over t'=0..489: V = r[10+t'] + GAMMA*E[11+t'] ; center; dot with out
  float sumV = 0.f;
  for (int q = 0; q < 490; ++q)
    sumV += (r_buf[(size_t)b * TSS + 10 + q] + 0.99f * E_buf[(size_t)b * TSS + 11 + q]);
  float Vbar = sumV / 490.f;
  float sVO = 0.f, sO = 0.f;
  for (int q = 0; q < 490; ++q) {
    float V = r_buf[(size_t)b * TSS + 10 + q] + 0.99f * E_buf[(size_t)b * TSS + 11 + q];
    float o = dout[(size_t)b * TSS + 10 + q];          // out_time_stock[b, 10+q, 0]
    sVO += V * o; sO += o;
  }
  float lossP = sVO - Vbar * sO;

  __shared__ float red[BSZ];
  // loss
  red[b] = lossP; __syncthreads();
  for (int off = 64; off > 0; off >>= 1) { if (b < off) red[b] += red[b + off]; __syncthreads(); }
  if (b == 0) dout[LOSS_OFF] = -red[0] / (128.f * 490.f);
  __syncthreads();
  // profit
  red[b] = E0; __syncthreads();
  for (int off = 64; off > 0; off >>= 1) { if (b < off) red[b] += red[b + off]; __syncthreads(); }
  if (b == 0) dout[PROFIT_OFF] = red[0] / 128.f;
  __syncthreads();
  // actual_reward
  red[b] = sum_r; __syncthreads();
  for (int off = 64; off > 0; off >>= 1) { if (b < off) red[b] += red[b + off]; __syncthreads(); }
  if (b == 0) dout[AR_OFF] = red[0] / 128.f;
}

// ---------------- workspace layout ----------------
constexpr size_t al256(size_t x) { return (x + 255) & ~(size_t)255; }
constexpr size_t OFF_WIH  = 0;
constexpr size_t OFF_WHH  = al256(OFF_WIH + (size_t)NGATE * 32 * 2);
constexpr size_t OFF_W1T  = al256(OFF_WHH + (size_t)NGATE * HSZ * 2);
constexpr size_t OFF_W2T  = al256(OFF_W1T + (size_t)NL1 * KMLP * 2);
constexpr size_t OFF_H0   = al256(OFF_W2T + (size_t)HSZ * NL1 * 2);
constexpr size_t OFF_H1   = al256(OFF_H0 + (size_t)MROWS * HSZ * 2);
constexpr size_t OFF_C    = al256(OFF_H1 + (size_t)MROWS * HSZ * 2);
constexpr size_t OFF_X1   = al256(OFF_C + (size_t)MROWS * HSZ * 4);
constexpr size_t OFF_X2   = al256(OFF_X1 + (size_t)BSZ * NL1 * 2);
constexpr size_t OFF_ST   = al256(OFF_X2 + (size_t)BSZ * HSZ * 4);
constexpr size_t OFF_OM   = al256(OFF_ST + (size_t)BSZ * SNS * 4);
constexpr size_t OFF_CASH = al256(OFF_OM + (size_t)BSZ * SNS * 4);
constexpr size_t OFF_ARAW = al256(OFF_CASH + (size_t)2 * BSZ * 4);
constexpr size_t OFF_R    = al256(OFF_ARAW + (size_t)BSZ * (TSS + 1) * 4);
constexpr size_t OFF_E    = al256(OFF_R + (size_t)BSZ * TSS * 4);
constexpr size_t WS_NEED  = al256(OFF_E + (size_t)BSZ * TSS * 4);

extern "C" void kernel_launch(void* const* d_in, const int* in_sizes, int n_in,
                              void* d_out, int out_size, void* d_ws, size_t ws_size,
                              hipStream_t stream) {
  const float* inputs = (const float*)d_in[0];
  const float* bp     = (const float*)d_in[1];
  const float* sp     = (const float*)d_in[2];
  const float* W_ih   = (const float*)d_in[3];
  const float* W_hh   = (const float*)d_in[4];
  const float* b_ih   = (const float*)d_in[5];
  const float* b_hh   = (const float*)d_in[6];
  const float* W1     = (const float*)d_in[7];
  const float* b1     = (const float*)d_in[8];
  const float* W2     = (const float*)d_in[9];
  const float* b2     = (const float*)d_in[10];
  const float* W3     = (const float*)d_in[11];
  const float* b3     = (const float*)d_in[12];
  const float* W4     = (const float*)d_in[13];
  const float* b4     = (const float*)d_in[14];
  float* dout = (float*)d_out;
  if (ws_size < WS_NEED) return;

  char* ws = (char*)d_ws;
  unsigned short* wih  = (unsigned short*)(ws + OFF_WIH);
  unsigned short* whh  = (unsigned short*)(ws + OFF_WHH);
  unsigned short* w1t  = (unsigned short*)(ws + OFF_W1T);
  unsigned short* w2t  = (unsigned short*)(ws + OFF_W2T);
  unsigned short* hbf0 = (unsigned short*)(ws + OFF_H0);
  unsigned short* hbf1 = (unsigned short*)(ws + OFF_H1);
  float* cbuf  = (float*)(ws + OFF_C);
  unsigned short* x1bf = (unsigned short*)(ws + OFF_X1);
  float* x2f   = (float*)(ws + OFF_X2);
  float* st    = (float*)(ws + OFF_ST);
  float* om    = (float*)(ws + OFF_OM);
  float* cash  = (float*)(ws + OFF_CASH);
  float* araw  = (float*)(ws + OFF_ARAW);
  float* rbuf  = (float*)(ws + OFF_R);
  float* ebuf  = (float*)(ws + OFF_E);

  // one-time prep (weights fit in the 192MB L2 and stay resident across the scan)
  k_prep_wih <<<(NGATE * 32 + 255) / 256, 256, 0, stream>>>(W_ih, wih);
  k_prep_whh <<<(NGATE * HSZ + 255) / 256, 256, 0, stream>>>(W_hh, whh);
  k_prep_w1t <<<((size_t)NL1 * KMLP + 255) / 256, 256, 0, stream>>>(W1, w1t);
  k_prep_w2t <<<(HSZ * NL1 + 255) / 256, 256, 0, stream>>>(W2, w2t);
  k_init_state<<<(MROWS * HSZ + 255) / 256, 256, 0, stream>>>(hbf0, cbuf, st, om, cash);

  // sequential scan: 4 launches per step, h double-buffered in bf16
  for (int t = 0; t < TSS; ++t) {
    const unsigned short* hin = (t & 1) ? hbf1 : hbf0;
    unsigned short*       hout = (t & 1) ? hbf0 : hbf1;
    k_lstm_step <<<512, 128, 0, stream>>>(inputs, wih, whh, b_ih, b_hh, hin, hout, cbuf, om, t);
    k_mlp1      <<<448, 128, 0, stream>>>(hout, w1t, b1, x1bf);
    k_mlp2      <<<64, 128, 0, stream>>>(x1bf, w2t, b2, x2f);
    k_head_trade<<<128, 32, 0, stream>>>(x2f, W3, b3, W4, b4, bp, sp, st, om, cash, araw, dout, t);
  }
  // post-pass: rewards, discounted returns, loss/profit/actual_reward, normalized asset
  k_post<<<1, 128, 0, stream>>>(bp, sp, araw, rbuf, ebuf, dout);
}
